// ViT_56281251447165
// MI455X (gfx1250) — matrix operand order repaired and checked by reading, not compiled, not run on hardware
//
#include <hip/hip_runtime.h>

// ---------------------------------------------------------------------------
// Types for gfx1250 WMMA
// ---------------------------------------------------------------------------
typedef __bf16 v16bf __attribute__((ext_vector_type(16)));
typedef __bf16 v8bf  __attribute__((ext_vector_type(8)));
typedef float  v8f   __attribute__((ext_vector_type(8)));

#define L_SEQ 2304
#define C_DIM 768
#define NH    12
#define HD    64
#define HS    48

__device__ __forceinline__ v8f wmma_bf16(v16bf a, v16bf b, v8f c) {
    return __builtin_amdgcn_wmma_f32_16x16x32_bf16(
        /*neg_a=*/false, a, /*neg_b=*/false, b,
        /*c_mod=*/(short)0, c, /*reuse_a=*/false, /*reuse_b=*/false);
}

// A-fragment (16x32, bf16): lane holds row m=lane&15; 16 contiguous K values
// starting at h*16 with h = lane>>4.
__device__ __forceinline__ v16bf ldA(const __bf16* base, int half) {
    v8bf lo = *(const v8bf*)(base + half * 16);
    v8bf hi = *(const v8bf*)(base + half * 16 + 8);
    return __builtin_shufflevector(lo, hi, 0,1,2,3,4,5,6,7,8,9,10,11,12,13,14,15);
}

// B-fragment (32x16, bf16): lane holds col n=lane&15; 16 contiguous K values
// starting at half*16.
__device__ __forceinline__ v16bf ldB(const __bf16* base, int half) {
    return *(const v16bf*)(base + half * 16);
}

// issue one per-lane 16B async global->LDS copy (ASYNCcnt path)
__device__ __forceinline__ void async_g2l_b128(unsigned lds_addr, const void* gaddr) {
    unsigned long long ga = (unsigned long long)(size_t)gaddr;
    asm volatile("global_load_async_to_lds_b128 %0, %1, off"
                 :: "v"(lds_addr), "v"(ga) : "memory");
}

// ---------------------------------------------------------------------------
// f32 -> bf16 cast
// ---------------------------------------------------------------------------
__global__ void cast_bf16_kernel(const float* __restrict__ in,
                                 __bf16* __restrict__ out, int n) {
    int i = blockIdx.x * blockDim.x + threadIdx.x;
    if (i < n) out[i] = (__bf16)in[i];
}

// ---------------------------------------------------------------------------
// LayerNorm over C=768, output bf16. One block (256 thr) per row.
// ---------------------------------------------------------------------------
__global__ __launch_bounds__(256)
void layernorm_kernel(const float* __restrict__ x, const float* __restrict__ w,
                      const float* __restrict__ b, __bf16* __restrict__ out) {
    const int row = blockIdx.x;
    const float* xr = x + (size_t)row * C_DIM;
    float vals[3], s = 0.f, s2 = 0.f;
    for (int i = 0; i < 3; ++i) {
        float v = xr[threadIdx.x + i * 256];
        vals[i] = v; s += v; s2 += v * v;
    }
    for (int off = 16; off > 0; off >>= 1) {
        s  += __shfl_xor(s,  off);
        s2 += __shfl_xor(s2, off);
    }
    __shared__ float sm[16];
    int wv = threadIdx.x >> 5;
    if ((threadIdx.x & 31) == 0) { sm[wv] = s; sm[8 + wv] = s2; }
    __syncthreads();
    s = 0.f; s2 = 0.f;
    for (int i = 0; i < 8; ++i) { s += sm[i]; s2 += sm[8 + i]; }
    float mu = s * (1.f / C_DIM);
    float var = s2 * (1.f / C_DIM) - mu * mu;
    float rstd = rsqrtf(var + 1e-5f);
    for (int i = 0; i < 3; ++i) {
        int c = threadIdx.x + i * 256;
        out[(size_t)row * C_DIM + c] = (__bf16)((vals[i] - mu) * rstd * w[c] + b[c]);
    }
}

// ---------------------------------------------------------------------------
// bf16 WMMA GEMM: out[M,N] = A[M,K] @ W[N,K]^T + bias (+GELU) (+resid)
// Wave computes a 32x64 tile (2 A-frags, 8 accumulators, 8 WMMA per K-step);
// block = 8 waves = 256 rows sharing one 64-col B strip staged through LDS
// with double-buffered global_load_async_to_lds_b128.
// Grid (N/64, M/256); M must be a multiple of 256 (true for all launches).
// ---------------------------------------------------------------------------
__global__ __launch_bounds__(256)
void gemm_bf16_kernel(const __bf16* __restrict__ A, const __bf16* __restrict__ W,
                      const float* __restrict__ bias, const float* __restrict__ resid,
                      float* __restrict__ outF, __bf16* __restrict__ outB,
                      int M, int N, int K, int act) {
    // two 64x32 bf16 tiles (4KB each), row stride 64B
    __shared__ __attribute__((aligned(128))) __bf16 Btile[2][64 * 32];

    const int lane = threadIdx.x & 31;
    const int wave = threadIdx.x >> 5;
    const int col  = lane & 15;
    const int half = lane >> 4;
    const int n0 = blockIdx.x * 64;
    const int m0 = (blockIdx.y * 8 + wave) * 32;

    // async-staging assignment: wave w copies B rows [w*8, w*8+8) of the tile,
    // 4 lanes per row, 16B per lane.
    const int brow = wave * 8 + (lane >> 2);
    const int bcol = (lane & 3) * 16;                       // byte offset in row
    const char* gsrc = (const char*)(W + (size_t)(n0 + brow) * K) + bcol;
    const unsigned lbase = (unsigned)(size_t)(&Btile[0][0]) + brow * 64 + bcol;
    const unsigned ldst0 = lbase;
    const unsigned ldst1 = lbase + 64 * 32 * 2;

    const __bf16* aRow0 = A + (size_t)(m0 + col) * K;
    const __bf16* aRow1 = A + (size_t)(m0 + 16 + col) * K;
    v8f c0 = {}, c1 = {}, c2 = {}, c3 = {};
    v8f c4 = {}, c5 = {}, c6 = {}, c7 = {};

    // prefetch K-tile 0 into buffer 0
    async_g2l_b128(ldst0, gsrc);

    for (int k = 0, s = 0; k < K; k += 32, ++s) {
        asm volatile("s_wait_asynccnt 0" ::: "memory");
        __syncthreads();                                    // tile s visible
        if (k + 32 < K)                                     // prefetch tile s+1
            async_g2l_b128((s & 1) ? ldst0 : ldst1,
                           gsrc + (size_t)(k + 32) * 2);

        const __bf16* bt = &Btile[s & 1][0];
        v16bf a0 = ldA(aRow0 + k, half);
        v16bf a1 = ldA(aRow1 + k, half);
        v16bf b0 = *(const v16bf*)(bt + (size_t)(col)      * 32 + half * 16);
        v16bf b1 = *(const v16bf*)(bt + (size_t)(16 + col) * 32 + half * 16);
        v16bf b2 = *(const v16bf*)(bt + (size_t)(32 + col) * 32 + half * 16);
        v16bf b3 = *(const v16bf*)(bt + (size_t)(48 + col) * 32 + half * 16);
        c0 = wmma_bf16(a0, b0, c0);
        c4 = wmma_bf16(a1, b0, c4);
        c1 = wmma_bf16(a0, b1, c1);
        c5 = wmma_bf16(a1, b1, c5);
        c2 = wmma_bf16(a0, b2, c2);
        c6 = wmma_bf16(a1, b2, c6);
        c3 = wmma_bf16(a0, b3, c3);
        c7 = wmma_bf16(a1, b3, c7);
    }

    for (int j = 0; j < 8; ++j) {
        v8f c = (j == 0) ? c0 : (j == 1) ? c1 : (j == 2) ? c2 : (j == 3) ? c3
              : (j == 4) ? c4 : (j == 5) ? c5 : (j == 6) ? c6 : c7;
        int n = n0 + (j & 3) * 16 + col;
        int mb = m0 + (j >> 2) * 16;
        float bv = bias ? bias[n] : 0.f;
        for (int i = 0; i < 8; ++i) {
            int m = mb + i + half * 8;
            float v = c[i] + bv;
            if (act) v = 0.5f * v * (1.f + erff(v * 0.70710678118f));  // exact GELU
            size_t off = (size_t)m * N + n;
            if (resid) v += resid[off];
            if (outF) outF[off] = v;
            if (outB) outB[off] = (__bf16)v;
        }
    }
}

// ---------------------------------------------------------------------------
// 2D axial RoPE + split qkv -> q_bf[NH,L,HD], k_bf[NH,L,HD], vT[NH,HD,L]
// ---------------------------------------------------------------------------
__global__ void rope_split_kernel(const float* __restrict__ qkv,
                                  __bf16* __restrict__ qb, __bf16* __restrict__ kb,
                                  __bf16* __restrict__ vt) {
    int idx = blockIdx.x * blockDim.x + threadIdx.x;
    if (idx >= NH * L_SEQ) return;
    int h = idx / L_SEQ, l = idx % L_SEQ;
    float tx = (float)(l % HS), ty = (float)(l / HS);
    const float* qr = qkv + (size_t)l * (3 * C_DIM) + h * HD;
    const float* kr = qr + C_DIM;
    const float* vr = qr + 2 * C_DIM;
    __bf16* qo = qb + ((size_t)h * L_SEQ + l) * HD;
    __bf16* ko = kb + ((size_t)h * L_SEQ + l) * HD;
    for (int j = 0; j < 32; ++j) {
        float fr  = powf(10000.f, -(float)(j & 15) * (1.f / 16.f));
        float ang = ((j < 16) ? tx : ty) * fr;
        float cs = cosf(ang), sn = sinf(ang);
        float qre = qr[2 * j], qim = qr[2 * j + 1];
        qo[2 * j]     = (__bf16)(qre * cs - qim * sn);
        qo[2 * j + 1] = (__bf16)(qre * sn + qim * cs);
        float kre = kr[2 * j], kim = kr[2 * j + 1];
        ko[2 * j]     = (__bf16)(kre * cs - kim * sn);
        ko[2 * j + 1] = (__bf16)(kre * sn + kim * cs);
    }
    __bf16* vo = vt + (size_t)h * HD * L_SEQ + l;
    for (int d = 0; d < HD; ++d) vo[(size_t)d * L_SEQ] = (__bf16)vr[d];
}

// ---------------------------------------------------------------------------
// Decomposed rel-pos bias tables
// ---------------------------------------------------------------------------
__global__ void relpos_bias_kernel(const __bf16* __restrict__ qb,
                                   const float* __restrict__ rph,
                                   const float* __restrict__ rpw,
                                   float* __restrict__ bh, float* __restrict__ bw) {
    int idx = blockIdx.x * blockDim.x + threadIdx.x;
    if (idx >= NH * L_SEQ * HS) return;
    int k  = idx % HS;
    int hl = idx / HS;
    int l  = hl % L_SEQ, h = hl / L_SEQ;
    int qh = l / HS, qw = l % HS;
    const __bf16* q  = qb + ((size_t)h * L_SEQ + l) * HD;
    const float*  rh = rph + (size_t)(qh - k + HS - 1) * HD;
    const float*  rw = rpw + (size_t)(qw - k + HS - 1) * HD;
    float sh = 0.f, sw = 0.f;
    for (int c = 0; c < HD; ++c) {
        float qq = (float)q[c];
        sh += qq * rh[c];
        sw += qq * rw[c];
    }
    bh[idx] = sh;
    bw[idx] = sw;
}

// ---------------------------------------------------------------------------
// Flash attention: one wave per (16-query tile, head). Online softmax over
// 72 steps of 32 keys. S and P@V via v_wmma_f32_16x16x32_bf16.
// ---------------------------------------------------------------------------
__global__ __launch_bounds__(32)
void flash_attn_kernel(const __bf16* __restrict__ qb, const __bf16* __restrict__ kbm,
                       const __bf16* __restrict__ vt, const float* __restrict__ bh,
                       const float* __restrict__ bw, __bf16* __restrict__ out) {
    __shared__ float biasS[16][96];                 // [m][kh(0..47) | 48+kw]
    __shared__ __attribute__((aligned(32))) __bf16 Pl[16][32];

    const int h    = blockIdx.y;
    const int q0   = blockIdx.x * 16;
    const int lane = threadIdx.x & 31;
    const int col  = lane & 15;
    const int half = lane >> 4;

    for (int t = lane; t < 16 * 96; t += 32) {
        int m = t / 96, j = t % 96;
        size_t base = ((size_t)h * L_SEQ + (q0 + m)) * HS;
        biasS[m][j] = (j < HS) ? bh[base + j] : bw[base + (j - HS)];
    }
    asm volatile("s_wait_dscnt 0" ::: "memory");

    const __bf16* qrow = qb + ((size_t)h * L_SEQ + q0 + col) * HD;
    v16bf a0 = ldA(qrow, half);
    v16bf a1 = ldA(qrow + 32, half);

    float rmax[8], rsum[8];
    v8f o0 = {}, o1 = {}, o2 = {}, o3 = {};
    for (int i = 0; i < 8; ++i) { rmax[i] = -1e30f; rsum[i] = 0.f; }

    for (int kb2 = 0; kb2 < L_SEQ; kb2 += 32) {
        v8f s0 = {}, s1 = {};
        {
            const __bf16* kr0 = kbm + ((size_t)h * L_SEQ + kb2 + col) * HD;
            v16bf b0 = ldB(kr0, half);
            v16bf b1 = ldB(kr0 + 32, half);
            s0 = wmma_bf16(a0, b0, s0);
            s0 = wmma_bf16(a1, b1, s0);
            const __bf16* kr1 = kr0 + (size_t)16 * HD;
            v16bf b2 = ldB(kr1, half);
            v16bf b3 = ldB(kr1 + 32, half);
            s1 = wmma_bf16(a0, b2, s1);
            s1 = wmma_bf16(a1, b3, s1);
        }
        int key0 = kb2 + col, key1 = key0 + 16;
        int kh0 = key0 / HS, kw0 = key0 % HS;
        int kh1 = key1 / HS, kw1 = key1 % HS;
        float p0[8], p1[8], tmax[8], tsum[8], alpha[8];
        for (int i = 0; i < 8; ++i) {
            int m = i + half * 8;
            float v0 = s0[i] * 0.125f + biasS[m][kh0] + biasS[m][48 + kw0];
            float v1 = s1[i] * 0.125f + biasS[m][kh1] + biasS[m][48 + kw1];
            p0[i] = v0; p1[i] = v1;
            tmax[i] = fmaxf(v0, v1);
        }
        for (int off = 1; off < 16; off <<= 1)
            for (int i = 0; i < 8; ++i)
                tmax[i] = fmaxf(tmax[i], __shfl_xor(tmax[i], off));
        for (int i = 0; i < 8; ++i) {
            float nm = fmaxf(rmax[i], tmax[i]);
            alpha[i] = __expf(rmax[i] - nm);
            rmax[i]  = nm;
            float e0 = __expf(p0[i] - nm), e1 = __expf(p1[i] - nm);
            p0[i] = e0; p1[i] = e1;
            tsum[i] = e0 + e1;
        }
        for (int off = 1; off < 16; off <<= 1)
            for (int i = 0; i < 8; ++i)
                tsum[i] += __shfl_xor(tsum[i], off);
        for (int i = 0; i < 8; ++i) {
            rsum[i] = rsum[i] * alpha[i] + tsum[i];
            o0[i] *= alpha[i]; o1[i] *= alpha[i];
            o2[i] *= alpha[i]; o3[i] *= alpha[i];
            int m = i + half * 8;
            Pl[m][col]      = (__bf16)p0[i];
            Pl[m][col + 16] = (__bf16)p1[i];
        }
        asm volatile("s_wait_dscnt 0" ::: "memory");
        v16bf pa = ldA(&Pl[col][0], half);
        const __bf16* vb = vt + ((size_t)(h * HD + col)) * L_SEQ + kb2 + half * 16;
        o0 = wmma_bf16(pa, *(const v16bf*)(vb),                       o0);
        o1 = wmma_bf16(pa, *(const v16bf*)(vb + (size_t)16 * L_SEQ), o1);
        o2 = wmma_bf16(pa, *(const v16bf*)(vb + (size_t)32 * L_SEQ), o2);
        o3 = wmma_bf16(pa, *(const v16bf*)(vb + (size_t)48 * L_SEQ), o3);
    }

    for (int i = 0; i < 8; ++i) {
        int m = i + half * 8;
        float inv = 1.f / rsum[i];
        size_t base = (size_t)(q0 + m) * C_DIM + h * HD + col;
        out[base]      = (__bf16)(o0[i] * inv);
        out[base + 16] = (__bf16)(o1[i] * inv);
        out[base + 32] = (__bf16)(o2[i] * inv);
        out[base + 48] = (__bf16)(o3[i] * inv);
    }
}

// ---------------------------------------------------------------------------
// Host-side orchestration
// ---------------------------------------------------------------------------
extern "C" void kernel_launch(void* const* d_in, const int* in_sizes, int n_in,
                              void* d_out, int out_size, void* d_ws, size_t ws_size,
                              hipStream_t stream) {
    const float* x     = (const float*)d_in[0];
    const float* ln1w  = (const float*)d_in[1];
    const float* ln1b  = (const float*)d_in[2];
    const float* qkvw  = (const float*)d_in[3];
    const float* qkvb  = (const float*)d_in[4];
    const float* rph   = (const float*)d_in[5];
    const float* rpw   = (const float*)d_in[6];
    const float* projw = (const float*)d_in[7];
    const float* projb = (const float*)d_in[8];
    const float* ln2w  = (const float*)d_in[9];
    const float* ln2b  = (const float*)d_in[10];
    const float* fc1w  = (const float*)d_in[11];
    const float* fc1b  = (const float*)d_in[12];
    const float* fc2w  = (const float*)d_in[13];
    const float* fc2b  = (const float*)d_in[14];
    float* outp = (float*)d_out;

    char* ws = (char*)d_ws;
    auto carve = [&](size_t bytes) -> char* {
        char* p = ws;
        ws += (bytes + 255) & ~(size_t)255;
        return p;
    };
    const size_t LC  = (size_t)L_SEQ * C_DIM;
    __bf16* hbf     = (__bf16*)carve(LC * 2);
    __bf16* qkvw_b  = (__bf16*)carve((size_t)3 * C_DIM * C_DIM * 2);
    __bf16* projw_b = (__bf16*)carve((size_t)C_DIM * C_DIM * 2);
    __bf16* fc1w_b  = (__bf16*)carve((size_t)4 * C_DIM * C_DIM * 2);
    __bf16* fc2w_b  = (__bf16*)carve((size_t)C_DIM * 4 * C_DIM * 2);
    float*  qkv_f   = (float*)carve((size_t)L_SEQ * 3 * C_DIM * 4);
    __bf16* q_bf    = (__bf16*)carve((size_t)NH * L_SEQ * HD * 2);
    __bf16* k_bf    = (__bf16*)carve((size_t)NH * L_SEQ * HD * 2);
    __bf16* v_t     = (__bf16*)carve((size_t)NH * HD * L_SEQ * 2);
    float*  bh      = (float*)carve((size_t)NH * L_SEQ * HS * 4);
    float*  bw      = (float*)carve((size_t)NH * L_SEQ * HS * 4);
    __bf16* attn_bf = (__bf16*)carve(LC * 2);
    float*  x2      = (float*)carve(LC * 4);
    __bf16* h2bf    = (__bf16*)carve(LC * 2);
    __bf16* gbf     = (__bf16*)carve((size_t)L_SEQ * 4 * C_DIM * 2);
    (void)ws_size; (void)n_in; (void)in_sizes; (void)out_size;

    auto cast = [&](const float* src, __bf16* dst, int n) {
        cast_bf16_kernel<<<(n + 255) / 256, 256, 0, stream>>>(src, dst, n);
    };
    cast(qkvw,  qkvw_b,  3 * C_DIM * C_DIM);
    cast(projw, projw_b, C_DIM * C_DIM);
    cast(fc1w,  fc1w_b,  4 * C_DIM * C_DIM);
    cast(fc2w,  fc2w_b,  C_DIM * 4 * C_DIM);

    layernorm_kernel<<<L_SEQ, 256, 0, stream>>>(x, ln1w, ln1b, hbf);

    gemm_bf16_kernel<<<dim3(3 * C_DIM / 64, L_SEQ / 256), 256, 0, stream>>>(
        hbf, qkvw_b, qkvb, nullptr, qkv_f, nullptr, L_SEQ, 3 * C_DIM, C_DIM, 0);

    rope_split_kernel<<<(NH * L_SEQ + 255) / 256, 256, 0, stream>>>(
        qkv_f, q_bf, k_bf, v_t);

    relpos_bias_kernel<<<(NH * L_SEQ * HS + 255) / 256, 256, 0, stream>>>(
        q_bf, rph, rpw, bh, bw);

    flash_attn_kernel<<<dim3(L_SEQ / 16, NH), 32, 0, stream>>>(
        q_bf, k_bf, v_t, bh, bw, attn_bf);

    gemm_bf16_kernel<<<dim3(C_DIM / 64, L_SEQ / 256), 256, 0, stream>>>(
        attn_bf, projw_b, projb, x, x2, nullptr, L_SEQ, C_DIM, C_DIM, 0);

    layernorm_kernel<<<L_SEQ, 256, 0, stream>>>(x2, ln2w, ln2b, h2bf);

    gemm_bf16_kernel<<<dim3(4 * C_DIM / 64, L_SEQ / 256), 256, 0, stream>>>(
        h2bf, fc1w_b, fc1b, nullptr, nullptr, gbf, L_SEQ, 4 * C_DIM, C_DIM, 1);

    gemm_bf16_kernel<<<dim3(C_DIM / 64, L_SEQ / 256), 256, 0, stream>>>(
        gbf, fc2w_b, fc2b, x2, outp, nullptr, L_SEQ, C_DIM, 4 * C_DIM, 0);
}
